// GKAN_Nodes_19559281066594
// MI455X (gfx1250) — compile-verified, matrix operand order checked by reading, not run on hardware
//
#include <hip/hip_runtime.h>
#include <hip/hip_bf16.h>
#include <cstdint>

// ---------- vector types ----------
typedef __attribute__((ext_vector_type(16))) __bf16          v16bf;
typedef __attribute__((ext_vector_type(16))) unsigned short  v16u;
typedef __attribute__((ext_vector_type(8)))  unsigned short  u16x8;
typedef __attribute__((ext_vector_type(4)))  unsigned short  u16x4;
typedef __attribute__((ext_vector_type(8)))  float           v8f;
typedef __attribute__((ext_vector_type(4)))  float           f32x4;

__device__ __forceinline__ unsigned short f2bf(float f) {
    union { float f; unsigned u; } v; v.f = f;
    unsigned r = v.u + 0x7FFFu + ((v.u >> 16) & 1u);   // round-to-nearest-even
    return (unsigned short)(r >> 16);
}

// =====================================================================
// Strided f32 -> bf16 cast, 4 elements/thread (cols, stride, off all %4==0)
// dst[r*dstStride + dstOff + c] = bf16(src[r*cols + c])
// =====================================================================
__global__ void cast_bf16_kernel(const float* __restrict__ src,
                                 unsigned short* __restrict__ dst,
                                 long total4, int cols, int dstStride, int dstOff) {
    long idx4 = (long)blockIdx.x * blockDim.x + threadIdx.x;
    if (idx4 >= total4) return;
    long flat = idx4 * 4;
    long r = flat / cols;
    int  c = (int)(flat - r * cols);
    f32x4 v = *(const f32x4*)(src + flat);
    u16x4 o;
#pragma unroll
    for (int j = 0; j < 4; ++j) o[j] = f2bf(v[j]);
    *(u16x4*)(dst + r * (long)dstStride + dstOff + c) = o;
}

__global__ void zero_f32_kernel(float* __restrict__ p, long n) {
    long idx = (long)blockIdx.x * blockDim.x + threadIdx.x;
    if (idx < n) p[idx] = 0.0f;
}

// =====================================================================
// Tiled bf16 WMMA GEMM:  C[M,N] = A[M,K] @ W[N,K]^T (+ bias[N])
// block = 128 threads (4 waves), 16 rows x 64 cols per block.
// A k-chunks staged into double-buffered LDS with CDNA5 async-to-LDS
// copies (ASYNCcnt); next tile's copy overlaps current tile's WMMA.
// =====================================================================
__global__ __launch_bounds__(128) void gemm_bf16_kernel(
    const unsigned short* __restrict__ A,   // M x K, bf16 bits
    const unsigned short* __restrict__ W,   // N x K, bf16 bits
    const float* __restrict__ bias,         // N or nullptr
    float* __restrict__ C,                  // M x N, f32
    int M, int N, int K) {
    __shared__ unsigned short sA[2][16 * 32];

    const int tid  = threadIdx.x;
    const int lane = tid & 31;
    const int wave = tid >> 5;
    const int m0   = blockIdx.x * 16;
    const int n0   = blockIdx.y * 64 + wave * 16;
    const bool active = (n0 < N);           // wave-uniform

    const int mrow  = lane & 15;
    const int kbase = (lane >> 4) * 8;      // ISA 16-bit A-matrix layout
    const int ncol  = n0 + (lane & 15);

    // async stage of one 16x32 bf16 tile: 64 lanes x 16 bytes
    auto issue = [&](int buf, int kc) {
        if (tid < 64) {                      // waves 0,1 (wave-uniform)
            int row = tid >> 2, chunk = tid & 3;
            const unsigned short* src = A + (size_t)(m0 + row) * K + kc + chunk * 8;
            unsigned ldsAddr = (unsigned)(uintptr_t)&sA[buf][row * 32 + chunk * 8];
            asm volatile("global_load_async_to_lds_b128 %0, %1, off"
                         :: "v"(ldsAddr), "v"(src) : "memory");
        }
    };

    const int nk = K / 32;
    issue(0, 0);
    v8f acc = {};
    for (int ik = 0; ik < nk; ++ik) {
        const int buf = ik & 1;
        if (ik + 1 < nk) {                   // prefetch next tile, then wait
            issue(buf ^ 1, (ik + 1) * 32);   // for the older copy only
            asm volatile("s_wait_asynccnt 1" ::: "memory");
        } else {
            asm volatile("s_wait_asynccnt 0" ::: "memory");
        }
        __syncthreads();
        if (active) {
            const int kc = ik * 32;
            u16x8 a_lo = *(const u16x8*)&sA[buf][mrow * 32 + kbase];
            u16x8 a_hi = *(const u16x8*)&sA[buf][mrow * 32 + 16 + kbase];
            const unsigned short* wrow = W + (size_t)ncol * K + kc;
            u16x8 b_lo = *(const u16x8*)&wrow[kbase];
            u16x8 b_hi = *(const u16x8*)&wrow[16 + kbase];
            v16u av, bv;
#pragma unroll
            for (int j = 0; j < 8; ++j) {
                av[j] = a_lo[j]; av[j + 8] = a_hi[j];
                bv[j] = b_lo[j]; bv[j + 8] = b_hi[j];
            }
            acc = __builtin_amdgcn_wmma_f32_16x16x32_bf16(
                false, __builtin_bit_cast(v16bf, av),
                false, __builtin_bit_cast(v16bf, bv),
                (short)0, acc, false, false);
        }
        __syncthreads();
    }
    if (active) {
        float bb = bias ? bias[ncol] : 0.0f;
        const int mo = (lane >> 4) * 8;     // C/D layout: lanes 16-31 -> M+8
#pragma unroll
        for (int r = 0; r < 8; ++r)
            C[(size_t)(m0 + r + mo) * N + ncol] = acc[r] + bb;
    }
}

// =====================================================================
// Fused edge kernel: per 16-edge tile
//   ea   = edge_attr[e,:] @ edge_W^T + edge_b   (WMMA, ea never hits HBM)
//   msg  = ea * xt[row[e], :]                   (L2-resident gather)
//   agg[col[e], :] += msg                       (f32 atomics in L2)
// =====================================================================
__global__ __launch_bounds__(256) void edge_fused_kernel(
    const float* __restrict__ edge_attr,     // E x 384 f32
    const unsigned short* __restrict__ Wb,   // Nout x 384 bf16
    const float* __restrict__ eb,            // Nout
    const int* __restrict__ rows,
    const int* __restrict__ cols,
    const float* __restrict__ xt,            // Nnodes x Nout f32
    float* __restrict__ agg,                 // Nnodes x Nout f32
    int Nout) {
    __shared__ unsigned short sEA[16 * 384]; // 12 KB bf16 edge tile
    __shared__ int sR[16], sC[16];

    const int tid  = threadIdx.x;
    const int lane = tid & 31;
    const int wave = tid >> 5;
    const long e0  = (long)blockIdx.x * 16;

    if (tid < 16)       sR[tid]      = rows[e0 + tid];
    else if (tid < 32)  sC[tid - 16] = cols[e0 + (tid - 16)];
    // stage + convert: 1536 float4 chunks, 6 per thread, vector LDS stores
    for (int c4 = tid; c4 < 1536; c4 += 256) {
        f32x4 v = *(const f32x4*)(edge_attr + e0 * 384 + (long)c4 * 4);
        u16x4 o;
#pragma unroll
        for (int j = 0; j < 4; ++j) o[j] = f2bf(v[j]);
        *(u16x4*)&sEA[c4 * 4] = o;
    }
    __syncthreads();

    const int mrow  = lane & 15;
    const int kbase = (lane >> 4) * 8;
    const int mo    = (lane >> 4) * 8;

    for (int n0 = wave * 16; n0 < Nout; n0 += 128) {   // wave-uniform loop
        const int ncol = n0 + (lane & 15);
        v8f acc = {};
#pragma unroll 4
        for (int kc = 0; kc < 384; kc += 32) {
            u16x8 a_lo = *(const u16x8*)&sEA[mrow * 384 + kc + kbase];
            u16x8 a_hi = *(const u16x8*)&sEA[mrow * 384 + kc + 16 + kbase];
            const unsigned short* wrow = Wb + (size_t)ncol * 384 + kc;
            u16x8 b_lo = *(const u16x8*)&wrow[kbase];
            u16x8 b_hi = *(const u16x8*)&wrow[16 + kbase];
            v16u av, bv;
#pragma unroll
            for (int j = 0; j < 8; ++j) {
                av[j] = a_lo[j]; av[j + 8] = a_hi[j];
                bv[j] = b_lo[j]; bv[j + 8] = b_hi[j];
            }
            acc = __builtin_amdgcn_wmma_f32_16x16x32_bf16(
                false, __builtin_bit_cast(v16bf, av),
                false, __builtin_bit_cast(v16bf, bv),
                (short)0, acc, false, false);
        }
        float bb = eb[ncol];
#pragma unroll
        for (int r = 0; r < 8; ++r) {
            int m = r + mo;                               // edge within tile
            float msg = (acc[r] + bb) * xt[(size_t)sR[m] * Nout + ncol];
            atomicAdd(&agg[(size_t)sC[m] * Nout + ncol], msg);
        }
    }
}

// =====================================================================
// KAN augmentation: Xaug[n, i*8 + 0]   = silu(agg[n,i])
//                   Xaug[n, i*8 + 1+c] = B_spline_c(agg[n,i]),  c = 0..6
// Uniform cubic grid: t_j = 0.5*(j-3) - 1, j = 0..10  (GRID=4, k=3)
// =====================================================================
__global__ void kan_prep_kernel(const float* __restrict__ agg,
                                unsigned short* __restrict__ Xaug,
                                long total) {
    long idx = (long)blockIdx.x * blockDim.x + threadIdx.x;
    if (idx >= total) return;
    float x = agg[idx];
    float s = x / (1.0f + __expf(-x));     // SiLU

    float t[11];
#pragma unroll
    for (int j = 0; j < 11; ++j) t[j] = 0.5f * (float)(j - 3) - 1.0f;
    float b[10];
#pragma unroll
    for (int j = 0; j < 10; ++j) b[j] = (x >= t[j] && x < t[j + 1]) ? 1.0f : 0.0f;
#pragma unroll
    for (int ord = 1; ord <= 3; ++ord) {
#pragma unroll
        for (int j = 0; j + ord < 10; ++j) {
            b[j] = (x - t[j]) / (t[j + ord] - t[j]) * b[j]
                 + (t[j + ord + 1] - x) / (t[j + ord + 1] - t[j + 1]) * b[j + 1];
        }
    }
    u16x8 o;
    o[0] = f2bf(s);
#pragma unroll
    for (int c = 0; c < 7; ++c) o[1 + c] = f2bf(b[c]);
    *(u16x8*)(Xaug + idx * 8) = o;         // one 16B store
}

// Interleave base_W (OxI) and spline_W (OxIx7) into Waug (O x 8I) bf16.
__global__ void build_waug_kernel(const float* __restrict__ baseW,
                                  const float* __restrict__ splineW,
                                  unsigned short* __restrict__ Waug,
                                  long total /* O*I */) {
    long idx = (long)blockIdx.x * blockDim.x + threadIdx.x;
    if (idx >= total) return;
    u16x8 o;
    o[0] = f2bf(baseW[idx]);
#pragma unroll
    for (int c = 0; c < 7; ++c) o[1 + c] = f2bf(splineW[idx * 7 + c]);
    *(u16x8*)(Waug + idx * 8) = o;
}

// =====================================================================
// Host-side orchestration
// =====================================================================
static inline long cdiv(long a, long b) { return (a + b - 1) / b; }

extern "C" void kernel_launch(void* const* d_in, const int* in_sizes, int n_in,
                              void* d_out, int out_size, void* d_ws, size_t ws_size,
                              hipStream_t stream) {
    const int F = 256, H = 256, ED = 384, OC = 32;
    const long N = in_sizes[0] / F;          // 50000
    const long E = in_sizes[1] / 2;          // 500000

    const float* x         = (const float*)d_in[0];
    const int*   ei        = (const int*)d_in[1];
    const int*   rows      = ei;
    const int*   cols      = ei + E;
    const float* edge_attr = (const float*)d_in[2];

    struct Layer { const float *nW, *nb, *eW, *eb, *bW, *sW; };
    Layer L[3];
    for (int l = 0; l < 3; ++l) {
        L[l].nW = (const float*)d_in[3 + 6 * l + 0];
        L[l].nb = (const float*)d_in[3 + 6 * l + 1];
        L[l].eW = (const float*)d_in[3 + 6 * l + 2];
        L[l].eb = (const float*)d_in[3 + 6 * l + 3];
        L[l].bW = (const float*)d_in[3 + 6 * l + 4];
        L[l].sW = (const float*)d_in[3 + 6 * l + 5];
    }

    // ---- workspace carve-out (256B aligned regions) ----
    size_t off = 0;
    auto carve = [&](size_t bytes) {
        size_t cur = off;
        off += (bytes + 255) & ~(size_t)255;
        return (char*)d_ws + cur;
    };
    unsigned short* xbf   = (unsigned short*)carve((size_t)N * 256 * 2);
    unsigned short* xcat  = (unsigned short*)carve((size_t)N * 768 * 2);
    float*          xt    = (float*)carve((size_t)N * 256 * 4);
    float*          agg   = (float*)carve((size_t)N * 256 * 4);
    float*          h1    = (float*)carve((size_t)N * 256 * 4);
    float*          h2    = (float*)carve((size_t)N * 256 * 4);
    unsigned short* Xaug  = (unsigned short*)carve((size_t)N * 2048 * 2);
    unsigned short* wnode = (unsigned short*)carve((size_t)256 * 256 * 2);
    unsigned short* wedge = (unsigned short*)carve((size_t)256 * 384 * 2);
    unsigned short* waug  = (unsigned short*)carve((size_t)256 * 2048 * 2);
    (void)ws_size; (void)n_in; (void)out_size;

    auto cast = [&](const float* s, unsigned short* d, long total, int c, int stride, int coff) {
        long total4 = total / 4;
        cast_bf16_kernel<<<dim3((unsigned)cdiv(total4, 256)), dim3(256), 0, stream>>>(
            s, d, total4, c, stride, coff);
    };
    auto gemm = [&](const unsigned short* A, const unsigned short* W, const float* bias,
                    float* C, int M, int Nn, int K) {
        gemm_bf16_kernel<<<dim3((unsigned)(M / 16), (unsigned)cdiv(Nn, 64)),
                           dim3(128), 0, stream>>>(A, W, bias, C, M, Nn, K);
    };

    auto run_layer = [&](const unsigned short* xin_bf, int Kin, const Layer& p,
                         int Nout, float* hout) {
        // weights -> bf16
        cast(p.nW, wnode, (long)Nout * Kin, Kin, Kin, 0);
        cast(p.eW, wedge, (long)Nout * ED, ED, ED, 0);
        // xt = x @ node_W^T + node_b
        gemm(xin_bf, wnode, p.nb, xt, (int)N, Nout, Kin);
        // agg = scatter-add of fused edge messages
        zero_f32_kernel<<<dim3((unsigned)cdiv(N * Nout, 256)), dim3(256), 0, stream>>>(
            agg, N * (long)Nout);
        edge_fused_kernel<<<dim3((unsigned)(E / 16)), dim3(256), 0, stream>>>(
            edge_attr, wedge, p.eb, rows, cols, xt, agg, Nout);
        // kan_linear(agg) as one GEMM with augmented K = 8*Nout
        build_waug_kernel<<<dim3((unsigned)cdiv((long)Nout * Nout, 256)), dim3(256), 0, stream>>>(
            p.bW, p.sW, waug, (long)Nout * Nout);
        kan_prep_kernel<<<dim3((unsigned)cdiv(N * Nout, 256)), dim3(256), 0, stream>>>(
            agg, Xaug, N * (long)Nout);
        gemm(Xaug, waug, nullptr, hout, (int)N, Nout, Nout * 8);
    };

    // ---- layer 0 ----
    cast(x, xbf, N * F, F, F, 0);
    run_layer(xbf, F, L[0], H, h1);
    // ---- layer 1 ----
    cast(h1, xbf, N * H, H, H, 0);
    run_layer(xbf, H, L[1], H, h2);
    // ---- output layer on concat [x, h1, h2] ----
    cast(x,  xcat, N * F, F, 768, 0);
    cast(h1, xcat, N * H, H, 768, 256);
    cast(h2, xcat, N * H, H, 768, 512);
    run_layer(xcat, F + 2 * H, L[2], OC, (float*)d_out);
}
